// RLST_20942260535339
// MI455X (gfx1250) — matrix operand order repaired and checked by reading, not compiled, not run on hardware
//
#include <hip/hip_runtime.h>

// ---------------- problem constants (match reference) ----------------
#define S_LEN   64
#define TT_LEN  64
#define BATCH   64
#define EMB     256
#define HID     512
#define VOC     10000
#define VO      10002          // VOC + 2
#define LDO     10016          // VO padded to multiple of 16
#define NT_OUT  626            // LDO / 16
#define TSTEPS  127            // S + TT - 1
#define NWG     128
#define NTHR    256
#define NWAVES  (NWG * NTHR / 32)   // 1024 wave-slots

#define OFF_WOUTS 0
#define N_WOUTS   ((size_t)TT_LEN * BATCH * VOC)        // 40,960,000
#define OFF_QU    (N_WOUTS)
#define OFF_QT    (N_WOUTS + (size_t)TSTEPS * BATCH)
#define OFF_AC    (N_WOUTS + 2u * (size_t)TSTEPS * BATCH)

// ---------------- WMMA fragment types ----------------
typedef __attribute__((ext_vector_type(16))) __bf16 v16bf;
typedef __attribute__((ext_vector_type(8)))  float  v8f;

union AFrag { unsigned u[8]; v16bf v; };
union BFrag { uint4    q[2]; v16bf v; };

__device__ __forceinline__ unsigned short f2bf(float f) {
  unsigned u = __float_as_uint(f);
  u += 0x7fffu + ((u >> 16) & 1u);      // round-to-nearest-even
  return (unsigned short)(u >> 16);
}

// A-matrix 16x32 bf16 fragment from a row-major [*,512] bf16 buffer.
// ISA 7.12.2: lanes 0-15 -> rows M=0..15 with K=0..7,16..23; lanes 16-31 -> K=8..15,24..31.
// Byte pattern per lane: 16 contiguous bytes at +2*ko and 16 at +32+2*ko -> two b128 loads.
__device__ __forceinline__ v16bf load_a(const unsigned short* A, int row, int kt, int lane) {
  const int ko = (lane >> 4) << 3;                 // 0 or 8
  const unsigned short* p = A + (size_t)row * HID + kt * 32;
  AFrag f;
#pragma unroll
  for (int v = 0; v < 4; ++v) {
    f.u[v]     = *(const unsigned*)(p + ko + 2 * v);        // K = ko+2v, ko+2v+1
    f.u[v + 4] = *(const unsigned*)(p + 16 + ko + 2 * v);   // K = 16+ko+2v, ...
  }
  return f.v;
}

// B-matrix fragment: pre-packed fragment-linear, 32 contiguous bytes per lane.
__device__ __forceinline__ v16bf load_b(const unsigned short* Wp, int nt, int kt, int lane) {
  const unsigned short* p = Wp + ((((size_t)nt * 16 + kt) * 32) + lane) * 16;
  BFrag f;
  f.q[0] = ((const uint4*)p)[0];
  f.q[1] = ((const uint4*)p)[1];
  return f.v;
}

// M-blocked n-tile: one wave computes all 4 m-tiles (full M=64) for one n-tile,
// loading each B fragment ONCE and feeding 4 WMMAs (4x less weight traffic).
__device__ __forceinline__ void gemm_nt(const unsigned short* A, const unsigned short* Wp,
                                        float* C, int nt, int ldc,
                                        const float* bias, int ncols, int lane) {
  v8f a0 = {}, a1 = {}, a2 = {}, a3 = {};
  const int r = lane & 15;
#pragma unroll 2
  for (int kt = 0; kt < 16; ++kt) {               // K = 512 = 16 * 32
    v16bf bm = load_b(Wp, nt, kt, lane);
    v16bf f0 = load_a(A,      r, kt, lane);
    v16bf f1 = load_a(A, 16 + r, kt, lane);
    v16bf f2 = load_a(A, 32 + r, kt, lane);
    v16bf f3 = load_a(A, 48 + r, kt, lane);
    a0 = __builtin_amdgcn_wmma_f32_16x16x32_bf16(false, f0, false, bm, (short)0, a0, false, false);
    a1 = __builtin_amdgcn_wmma_f32_16x16x32_bf16(false, f1, false, bm, (short)0, a1, false, false);
    a2 = __builtin_amdgcn_wmma_f32_16x16x32_bf16(false, f2, false, bm, (short)0, a2, false, false);
    a3 = __builtin_amdgcn_wmma_f32_16x16x32_bf16(false, f3, false, bm, (short)0, a3, false, false);
  }
  const int col = nt * 16 + r;
  if (col < ncols) {
    const float bv = bias ? bias[col] : 0.0f;
    const int r0 = (lane >> 4) << 3;              // C/D layout: lanes 16-31 hold M=8..15
#pragma unroll
    for (int rr = 0; rr < 8; ++rr) {
      C[(size_t)(r0 + rr)      * ldc + col] = a0[rr] + bv;
      C[(size_t)(r0 + rr + 16) * ldc + col] = a1[rr] + bv;
      C[(size_t)(r0 + rr + 32) * ldc + col] = a2[rr] + bv;
      C[(size_t)(r0 + rr + 48) * ldc + col] = a3[rr] + bv;
    }
  }
}

// ---------------- grid-wide barrier (all 128 WGs co-resident) ----------------
// Sense-reversal with alternating counters (bar[0]/bar[2]) + generation bar[1].
__device__ __forceinline__ void grid_barrier(unsigned* bar) {
  __syncthreads();
  __threadfence();                                 // release (every wave flushes)
  if (threadIdx.x == 0) {
    unsigned gen = atomicAdd(&bar[1], 0u);
    unsigned* cnt = &bar[(gen & 1u) ? 2 : 0];
    if (atomicAdd(cnt, 1u) == NWG - 1u) {
      atomicExch(cnt, 0u);
      __threadfence();
      atomicAdd(&bar[1], 1u);
    } else {
      while (atomicAdd(&bar[1], 0u) == gen) __builtin_amdgcn_s_sleep(2);
    }
  }
  __syncthreads();
  __threadfence();                                 // acquire (invalidate per-CU caches)
}

// ---------------- weight packing: f32 row-major -> bf16 B-fragment order ----------------
__global__ void k_pack(const float* __restrict__ W, unsigned short* __restrict__ dst,
                       int N, int NT) {
  const long total = (long)NT * 16 * 512;          // NT * KT(16) * 32lanes * 16elems
  for (long idx = (long)blockIdx.x * blockDim.x + threadIdx.x; idx < total;
       idx += (long)gridDim.x * blockDim.x) {
    const int e    = (int)(idx & 15);
    const int lane = (int)((idx >> 4) & 31);
    const int kt   = (int)((idx >> 9) & 15);
    const int nt   = (int)(idx >> 13);
    const int k = kt * 32 + ((lane >> 4) << 3) + (e & 7) + ((e >> 3) << 4);   // lanes 0-15: K=0..7,16..23; 16-31: K=8..15,24..31
    const int n = nt * 16 + (lane & 15);
    dst[idx] = (n < N) ? f2bf(W[(size_t)k * N + n]) : (unsigned short)0;
  }
}

__global__ void k_init(int* iA, int* jA, int* termA, int* acnt, unsigned* bar) {
  const int t = threadIdx.x;
  if (t < BATCH) { iA[t] = 0; jA[t] = 0; termA[t] = 0; }
  if (t < 4) { acnt[t] = 0; bar[t] = 0u; }
}

// ---------------- persistent episode kernel ----------------
struct EpArgs {
  const int *src, *trg;
  const float *src_emb, *trg_emb, *bih, *bhh, *bout;
  const unsigned short *wih_p, *whh_p, *wout_p;
  unsigned short *xb16, *hb16;
  float *h, *gx, *gh, *outb;
  float *reward;
  int *flags, *iA, *jA, *termA, *action, *acnt;
  unsigned *bar;
  float *out;                    // d_out
};

__global__ __launch_bounds__(NTHR) void k_episode(EpArgs E) {
  const int tid  = threadIdx.x;
  const int b    = blockIdx.x;                    // blocks 0..63 own batch row b in phase A
  const int gtid = b * NTHR + tid;
  const int wid  = gtid >> 5;
  const int lane = gtid & 31;

  __shared__ float s_val[NTHR];
  __shared__ int   s_idx[NTHR];
  __shared__ int   s_b[4];

  // ---- per-launch init: zero wouts region of d_out, zero h / hb16, gather x for step -1
  {
    float4 z; z.x = z.y = z.z = z.w = 0.0f;
    float4* o4 = (float4*)(E.out + OFF_WOUTS);
    for (size_t i = gtid; i < N_WOUTS / 4; i += (size_t)NWG * NTHR) o4[i] = z;
    for (int i = gtid; i < BATCH * HID; i += NWG * NTHR) { E.h[i] = 0.0f; E.hb16[i] = 0; }
    if (b < BATCH) {
      const int inp = E.src[b];                   // src[0][b], wo = TRG_NULL(3)
      for (int c = tid; c < 2 * EMB; c += NTHR)
        E.xb16[(size_t)b * HID + c] =
          f2bf(c < EMB ? E.src_emb[(size_t)inp * EMB + c]
                       : E.trg_emb[(size_t)3 * EMB + (c - EMB)]);
    }
  }
  grid_barrier(E.bar);

  for (int t = -1; t < TSTEPS; ++t) {
    if (t >= 0) {
      // ================= Phase A : per-row reductions + control flow =================
      if (b < BATCH) {
        const float* orow = E.outb + (size_t)b * LDO;
        // argmax over vocab (first occurrence on ties)
        float best = -3.4e38f; int bidx = 0;
        for (int c = tid; c < VOC; c += NTHR) {
          float v = orow[c];
          if (v > best) { best = v; bidx = c; }
        }
        s_val[tid] = best; s_idx[tid] = bidx; __syncthreads();
        for (int st = NTHR / 2; st > 0; st >>= 1) {
          if (tid < st) {
            float v2 = s_val[tid + st]; int i2 = s_idx[tid + st];
            if (v2 > s_val[tid] || (v2 == s_val[tid] && i2 < s_idx[tid])) {
              s_val[tid] = v2; s_idx[tid] = i2;
            }
          }
          __syncthreads();
        }
        const float mx = s_val[0];
        const int wofull = s_idx[0];
        __syncthreads();
        // sum exp(logit - mx)
        float ls = 0.0f;
        for (int c = tid; c < VOC; c += NTHR) ls += __expf(orow[c] - mx);
        s_val[tid] = ls; __syncthreads();
        for (int st = NTHR / 2; st > 0; st >>= 1) {
          if (tid < st) s_val[tid] += s_val[tid + st];
          __syncthreads();
        }
        if (tid == 0) {
          const float sumexp = s_val[0];
          const int act  = E.action[b];
          const int term = E.termA[b];
          const int ii = E.iA[b], jj = E.jA[b];
          const float qsel = orow[VOC + act];
          E.out[OFF_QU + (size_t)t * BATCH + b] = term ? 0.0f : qsel;
          const int reading = (!term) && (act == 0);
          const int writing = (!term) && (act == 1);
          if (reading) atomicAdd(&E.acnt[0], 1);
          if (writing) atomicAdd(&E.acnt[1], 1);
          const int just_term = writing && (E.trg[jj * BATCH + b] == 2);   // TRG_EOS
          const int naughty   = reading && (E.src[ii * BATCH + b] == 2);   // SRC_EOS
          int i_new = ii + ((reading && !naughty) ? 1 : 0);
          if (i_new > S_LEN - 1) i_new = S_LEN - 1;
          const int old_j = jj;
          int j_new = jj + (writing ? 1 : 0);
          if (j_new > TT_LEN - 1) j_new = TT_LEN - 1;
          const int wo_tok = reading ? 3 : wofull;                          // TRG_NULL
          const int tgt = E.trg[old_j * BATCH + b];
          const float rw = (tgt == 1) ? 0.0f
                                      : (orow[tgt] - mx - __logf(sumexp));  // reward = logp[tgt]
          int inp_tok = E.src[i_new * BATCH + b];
          if (writing) inp_tok = 3;                                         // SRC_NULL
          if (naughty) inp_tok = 2;                                         // SRC_EOS
          E.iA[b] = i_new; E.jA[b] = j_new; E.termA[b] = term || just_term;
          E.flags[b]  = term | (reading << 1) | (naughty << 2) | (just_term << 3);
          E.reward[b] = rw;
          s_b[0] = writing; s_b[1] = old_j; s_b[2] = inp_tok; s_b[3] = wo_tok;
        }
        __syncthreads();
        if (s_b[0]) {                               // wouts[old_j, b, :] = logits
          const size_t base = ((size_t)s_b[1] * BATCH + b) * VOC;
          for (int c = tid; c < VOC; c += NTHR) E.out[OFF_WOUTS + base + c] = orow[c];
        }
        {
          const int inp = s_b[2], wo = s_b[3];
          for (int c = tid; c < 2 * EMB; c += NTHR)
            E.xb16[(size_t)b * HID + c] =
              f2bf(c < EMB ? E.src_emb[(size_t)inp * EMB + c]
                           : E.trg_emb[(size_t)wo * EMB + (c - EMB)]);
        }
      }
      grid_barrier(E.bar);
    }

    // ===== GEMM1 : gx = x@Wih ; gh = h@Whh  (192 blocked n-tile units) =====
    for (int g = wid; g < 2 * 96; g += NWAVES) {
      const int m = (g >= 96);
      const int nt = g - (m ? 96 : 0);
      gemm_nt(m ? E.hb16 : E.xb16, m ? E.whh_p : E.wih_p, m ? E.gh : E.gx,
              nt, 3 * HID, nullptr, 3 * HID, lane);
    }
    grid_barrier(E.bar);

    // ================= GRU elementwise -> h2 (f32 + bf16) =================
    for (int e = gtid; e < BATCH * HID; e += NWG * NTHR) {
      const int bb = e >> 9, c = e & (HID - 1);
      const size_t g0 = (size_t)bb * 3 * HID + c;
      const float gr = (E.gx[g0] + E.bih[c]) + (E.gh[g0] + E.bhh[c]);
      const float gz = (E.gx[g0 + HID] + E.bih[c + HID]) + (E.gh[g0 + HID] + E.bhh[c + HID]);
      const float r = 1.0f / (1.0f + __expf(-gr));
      const float z = 1.0f / (1.0f + __expf(-gz));
      const float gxn = E.gx[g0 + 2 * HID] + E.bih[c + 2 * HID];
      const float ghn = E.gh[g0 + 2 * HID] + E.bhh[c + 2 * HID];
      const float n = tanhf(gxn + r * ghn);
      const float h2 = (1.0f - z) * n + z * E.h[e];
      E.h[e] = h2;
      E.hb16[e] = f2bf(h2);
    }
    grid_barrier(E.bar);

    // ===== GEMM2 : out = h2@Wout + bout  (626 blocked n-tile units, 1 round) =====
    for (int nt = wid; nt < NT_OUT; nt += NWAVES)
      gemm_nt(E.hb16, E.wout_p, E.outb, nt, LDO, E.bout, VO, lane);
    grid_barrier(E.bar);

    // ================= Phase F : action / Q-target =================
    if (b == 0 && tid < BATCH) {
      const int bb = tid;
      const float o0 = E.outb[(size_t)bb * LDO + VOC];
      const float o1 = E.outb[(size_t)bb * LDO + VOC + 1];
      const int   act2 = (o1 > o0) ? 1 : 0;        // argmax, first on tie
      const float nbv  = fmaxf(o0, o1);
      if (t >= 0) {
        const int fl = E.flags[bb];
        const float rw = E.reward[bb];
        const int term_old = fl & 1, reading = (fl >> 1) & 1;
        const int naughty = (fl >> 2) & 1, just_term = (fl >> 3) & 1;
        float qt = rw + 0.9f * nbv;
        if (term_old)           qt = 0.0f;
        if (reading)            qt = nbv;
        if (reading && naughty) qt = 0.9f * nbv;
        if (just_term)          qt = rw;
        if (naughty)            qt -= 7.0f;        // M
        E.out[OFF_QT + (size_t)t * BATCH + bb] = qt;
      }
      E.action[bb] = act2;
    }
    grid_barrier(E.bar);
  }

  if (b == 0 && tid < 3) E.out[OFF_AC + tid] = (float)E.acnt[tid];
}

// ---------------- host side ----------------
extern "C" void kernel_launch(void* const* d_in, const int* in_sizes, int n_in,
                              void* d_out, int out_size, void* d_ws, size_t ws_size,
                              hipStream_t stream) {
  const int*   src     = (const int*)d_in[0];
  const int*   trg     = (const int*)d_in[1];
  const float* src_emb = (const float*)d_in[2];
  const float* trg_emb = (const float*)d_in[3];
  const float* Wih     = (const float*)d_in[4];
  const float* Whh     = (const float*)d_in[5];
  const float* bih     = (const float*)d_in[6];
  const float* bhh     = (const float*)d_in[7];
  const float* Wout    = (const float*)d_in[8];
  const float* bout    = (const float*)d_in[9];

  char* ws = (char*)d_ws;
  size_t off = 0;
  auto carve = [&](size_t bytes) -> char* {
    char* p = ws + off;
    off = (off + bytes + 255) & ~(size_t)255;
    return p;
  };
  unsigned short* wih_p  = (unsigned short*)carve((size_t)96 * 16 * 512 * 2);
  unsigned short* whh_p  = (unsigned short*)carve((size_t)96 * 16 * 512 * 2);
  unsigned short* wout_p = (unsigned short*)carve((size_t)NT_OUT * 16 * 512 * 2);
  unsigned short* xb16   = (unsigned short*)carve((size_t)BATCH * HID * 2);
  unsigned short* hb16   = (unsigned short*)carve((size_t)BATCH * HID * 2);
  float* h    = (float*)carve((size_t)BATCH * HID * 4);
  float* gx   = (float*)carve((size_t)BATCH * 3 * HID * 4);
  float* gh   = (float*)carve((size_t)BATCH * 3 * HID * 4);
  float* outb = (float*)carve((size_t)BATCH * LDO * 4);
  float* reward = (float*)carve(BATCH * 4);
  int* flags  = (int*)carve(BATCH * 4);
  int* iA     = (int*)carve(BATCH * 4);
  int* jA     = (int*)carve(BATCH * 4);
  int* termA  = (int*)carve(BATCH * 4);
  int* action = (int*)carve(BATCH * 4);
  int* acnt   = (int*)carve(4 * 4);
  unsigned* bar = (unsigned*)carve(4 * 4);

  k_init<<<1, 64, 0, stream>>>(iA, jA, termA, acnt, bar);
  k_pack<<<1024, 256, 0, stream>>>(Wih,  wih_p,  3 * HID, 96);
  k_pack<<<1024, 256, 0, stream>>>(Whh,  whh_p,  3 * HID, 96);
  k_pack<<<2048, 256, 0, stream>>>(Wout, wout_p, VO, NT_OUT);

  EpArgs E;
  E.src = src; E.trg = trg; E.src_emb = src_emb; E.trg_emb = trg_emb;
  E.bih = bih; E.bhh = bhh; E.bout = bout;
  E.wih_p = wih_p; E.whh_p = whh_p; E.wout_p = wout_p;
  E.xb16 = xb16; E.hb16 = hb16;
  E.h = h; E.gx = gx; E.gh = gh; E.outb = outb;
  E.reward = reward; E.flags = flags; E.iA = iA; E.jA = jA;
  E.termA = termA; E.action = action; E.acnt = acnt; E.bar = bar;
  E.out = (float*)d_out;

  k_episode<<<NWG, NTHR, 0, stream>>>(E);
}
